// RecurrentStateSpaceModel_27341761806716
// MI455X (gfx1250) — matrix operand order repaired and checked by reading, not compiled, not run on hardware
//
#include <hip/hip_runtime.h>
#include <hip/hip_bf16.h>
#include <math.h>

// ---------------------------------------------------------------------------
// RSSM forward for MI455X (gfx1250, wave32).
// - All GEMMs (convs via implicit im2col, MLPs, GRU mats) on
//   v_wmma_f32_16x16x32_f16, f32 accumulation.
// - Activations & weights f16 in global; weights pre-swizzled to fragment
//   order so B-tiles are contiguous 2KB blocks moved with
//   global_load_async_to_lds_b128 (ASYNCcnt).
// - A-tiles: 8-aligned K-chunks map to 16 contiguous LDS bytes in fragment
//   order -> one global_load_b128 + one ds_store_b128 per thread per tile.
// - Fragment reads: ds_load_b128 x2 per WMMA operand.
// ---------------------------------------------------------------------------

typedef __attribute__((ext_vector_type(16))) _Float16 v16h;
typedef __attribute__((ext_vector_type(8)))  float    v8f;

#define TILE 32
enum { ACT_NONE = 0, ACT_RELU = 1, ACT_SP = 2 };   // SP = softplus + MIN_STD
#define MIN_STD 0.1f

#if defined(__HIP_DEVICE_COMPILE__) && \
    __has_builtin(__builtin_amdgcn_global_load_async_to_lds_b128) && \
    __has_builtin(__builtin_amdgcn_s_wait_asynccnt)
#define USE_ASYNC 1
typedef int i32x4 __attribute__((__vector_size__(16)));   // matches builtin param type
#else
#define USE_ASYNC 0
#endif

__device__ __forceinline__ float apply_act(float x, int ACT) {
    if (ACT == ACT_RELU) return x > 0.f ? x : 0.f;
    if (ACT == ACT_SP) {
        float sp = (x > 20.f) ? x : log1pf(__expf(x));
        return sp + MIN_STD;
    }
    return x;
}

// ---------------------------------------------------------------------------
// Fragment order used here (wave32; A/B use contiguous k, not the ISA 7.12.2 table order):
//  A 16x32 f16: lane = (m&15) + 16*(k>>4), elem j = k&15
//    -> for an 8-aligned chunk q (k = 8q..8q+7): lane = (m&15)+16*(q>>1),
//       j = (q&1)*8 + (0..7) contiguous.
//  B 32x16 f16: lane = (n&15) + 16*(k>>4), elem j = k&15
//  C 16x16 f32: elem v -> m = v + 8*(lane>>4), n = lane&15
// LDS tiles: [tile16][lane(32)][j(16)]; lane's v16h operand contiguous (32B).
// Packed weights: Bsw[kt][nt][lane][j], kt = ceil(K/32), nt = 2*ceil(N/32).
// ---------------------------------------------------------------------------

// convCin > 0 remaps conv im2col K to channel-minor order:
//   logical k = (kh*4+kw)*Cin + ci  ->  source k' = ci*16 + (kh*4+kw)
__global__ void pack_wsw_k(const float* __restrict__ src, _Float16* __restrict__ dst,
                           int K, int N, int transposed, int convCin) {
    const int KT = (K + 31) >> 5;
    const int NT = ((N + 31) >> 5) * 2;
    const long long total = (long long)KT * NT * 512;
    long long i = (long long)blockIdx.x * blockDim.x + threadIdx.x;
    if (i >= total) return;
    const int j    = (int)(i & 15);
    const int lane = (int)((i >> 4) & 31);
    const long long blk = i >> 9;
    const int nt = (int)(blk % NT);
    const int kt = (int)(blk / NT);
    const int k = kt * 32 + (lane >> 4) * 16 + j;
    const int n = nt * 16 + (lane & 15);
    float v = 0.f;
    if (k < K && n < N) {
        int sk = k;
        if (convCin > 0) sk = (k % convCin) * 16 + (k / convCin);
        v = transposed ? src[(long long)n * K + sk] : src[(long long)sk * N + n];
    }
    dst[i] = (_Float16)v;
}

// ---------------------------------------------------------------------------
// GEMM: C[m,n] = act(A[m,:] @ W[:,n] + bias[n]).  A f16, (m,k)->A[m*sA+k].
// Requires: M % 32 == 0, gridDim.y == M/32, sA % 8 == 0, A base 16B aligned.
// Block = 128 thr = 4 waves (2x2 of 16x16 WMMA tiles), block tile 32x32.
// ---------------------------------------------------------------------------
template <int ACT, typename TC>
__global__ __launch_bounds__(128)
void wmma_gemm(const _Float16* __restrict__ A, long long sA,
               const _Float16* __restrict__ Bsw,
               const float* __restrict__ bias,
               TC* __restrict__ C, long long sC,
               int N, int K) {
    __shared__ alignas(32) _Float16 As[2][32][16];
    __shared__ alignas(32) _Float16 Bs[2][32][16];

    const int tid  = threadIdx.x;
    const int lane = tid & 31;
    const int wave = tid >> 5;
    const int wm   = (wave & 1) * 16;
    const int wn   = (wave >> 1) * 16;
    const int bm   = blockIdx.y * TILE;
    const int bn   = blockIdx.x * TILE;

    const int KT = (K + 31) >> 5;
    const int NT = ((N + 31) >> 5) * 2;

    // A-loader: one (row, 8-col chunk) per thread. 128 thr = 32 rows x 4 chunks.
    const int r = tid >> 2;           // tile row
    const int q = tid & 3;            // 8-wide K chunk
    const _Float16* arow = A + (long long)(bm + r) * sA;
    _Float16* adst = &As[r >> 4][(r & 15) + (q >> 1) * 16][(q & 1) * 8];

    // B-loader: contiguous 2KB fragment-ordered block per K-tile; 16B/thread.
    const _Float16* bsrc0 = Bsw + ((size_t)(bn >> 4) << 9) + (size_t)tid * 8;
    const size_t    bstep = (size_t)NT << 9;
    _Float16*       bdst  = &Bs[0][0][0] + (size_t)tid * 8;

    v8f acc = {0.f, 0.f, 0.f, 0.f, 0.f, 0.f, 0.f, 0.f};

    for (int kt = 0; kt < KT; ++kt) {
        const _Float16* bsrc = bsrc0 + (size_t)kt * bstep;
#if USE_ASYNC
        __builtin_amdgcn_global_load_async_to_lds_b128((i32x4*)bsrc, (i32x4*)bdst, 0, 0);
#else
        *(uint4*)bdst = *(const uint4*)bsrc;
#endif
        const int k0 = kt * TILE + q * 8;
        if (k0 + 8 <= K) {
            *(uint4*)adst = *(const uint4*)(arow + k0);            // b128 -> b128
        } else if (k0 < K) {                                       // rare K tail
#pragma unroll
            for (int e = 0; e < 8; ++e)
                adst[e] = (k0 + e < K) ? arow[k0 + e] : (_Float16)0.f;
        } else {
            *(uint4*)adst = make_uint4(0u, 0u, 0u, 0u);
        }
#if USE_ASYNC
        __builtin_amdgcn_s_wait_asynccnt(0);
#endif
        __syncthreads();
        const v16h af = *(const v16h*)As[wm >> 4][lane];
        const v16h bf = *(const v16h*)Bs[wn >> 4][lane];
        acc = __builtin_amdgcn_wmma_f32_16x16x32_f16(false, af, false, bf,
                                                     (short)0, acc, false, false);
        __syncthreads();
    }

    const int half = lane >> 4, l16 = lane & 15;
#pragma unroll
    for (int v = 0; v < 8; ++v) {
        const int m = bm + wm + v + 8 * half;
        const int n = bn + wn + l16;
        if (n < N)
            C[(long long)m * sC + n] = (TC)apply_act(acc[v] + bias[n], ACT);
    }
}

// ---------------------------------------------------------------------------
// Implicit-GEMM conv (stride 2, 4x4, VALID), NHWC f16 in/out, relu.
// M = B*OH*OW (mult of 32), K = Cin*16 (channel-minor im2col), N = Cout.
// Cin % 8 == 0 -> every 8-chunk is contiguous in NHWC memory (b128 path);
// otherwise scalar fallback (only conv1, Cin = 3).
// ---------------------------------------------------------------------------
__global__ __launch_bounds__(128)
void wmma_conv(const _Float16* __restrict__ in,
               const _Float16* __restrict__ Wsw,
               const float* __restrict__ bias,
               _Float16* __restrict__ out,
               int Cin, int H, int W, int Cout, int OH, int OW) {
    __shared__ alignas(32) _Float16 As[2][32][16];
    __shared__ alignas(32) _Float16 Bs[2][32][16];

    const int tid  = threadIdx.x;
    const int lane = tid & 31;
    const int wave = tid >> 5;
    const int wm   = (wave & 1) * 16;
    const int wn   = (wave >> 1) * 16;
    const int bm   = blockIdx.y * TILE;
    const int bn   = blockIdx.x * TILE;

    const int K  = Cin * 16;
    const int KT = (K + 31) >> 5;
    const int NT = ((Cout + 31) >> 5) * 2;
    const bool vec8 = (Cin & 7) == 0;

    // One (row, chunk) per thread; im2col pixel decomposition hoisted.
    const int r = tid >> 2;
    const int q = tid & 3;
    const int gm  = bm + r;
    const int b   = gm / (OH * OW);
    const int rem = gm % (OH * OW);
    const int oh  = rem / OW, ow = rem % OW;
    const _Float16* abase = in + (((long long)b * H + oh * 2) * W + ow * 2) * Cin;
    _Float16* adst = &As[r >> 4][(r & 15) + (q >> 1) * 16][(q & 1) * 8];

    const _Float16* bsrc0 = Wsw + ((size_t)(bn >> 4) << 9) + (size_t)tid * 8;
    const size_t    bstep = (size_t)NT << 9;
    _Float16*       bdst  = &Bs[0][0][0] + (size_t)tid * 8;

    v8f acc = {0.f, 0.f, 0.f, 0.f, 0.f, 0.f, 0.f, 0.f};

    for (int kt = 0; kt < KT; ++kt) {
        const _Float16* bsrc = bsrc0 + (size_t)kt * bstep;
#if USE_ASYNC
        __builtin_amdgcn_global_load_async_to_lds_b128((i32x4*)bsrc, (i32x4*)bdst, 0, 0);
#else
        *(uint4*)bdst = *(const uint4*)bsrc;
#endif
        const int k0 = kt * TILE + q * 8;
        if (k0 < K) {
            if (vec8) {       // chunk lies inside one (kh,kw) cell, contiguous
                const int cell = k0 / Cin, ci0 = k0 % Cin;
                const int kh = cell >> 2, kw = cell & 3;
                *(uint4*)adst =
                    *(const uint4*)(abase + ((long long)kh * W + kw) * Cin + ci0);
            } else {
#pragma unroll
                for (int e = 0; e < 8; ++e) {
                    const int k = k0 + e;
                    _Float16 v = (_Float16)0.f;
                    if (k < K) {
                        const int cell = k / Cin, ci = k % Cin;
                        const int kh = cell >> 2, kw = cell & 3;
                        v = abase[((long long)kh * W + kw) * Cin + ci];
                    }
                    adst[e] = v;
                }
            }
        } else {
            *(uint4*)adst = make_uint4(0u, 0u, 0u, 0u);
        }
#if USE_ASYNC
        __builtin_amdgcn_s_wait_asynccnt(0);
#endif
        __syncthreads();
        const v16h af = *(const v16h*)As[wm >> 4][lane];
        const v16h bf = *(const v16h*)Bs[wn >> 4][lane];
        acc = __builtin_amdgcn_wmma_f32_16x16x32_f16(false, af, false, bf,
                                                     (short)0, acc, false, false);
        __syncthreads();
    }

    const int half = lane >> 4, l16 = lane & 15;
#pragma unroll
    for (int v = 0; v < 8; ++v) {
        const int m = bm + wm + v + 8 * half;
        const int n = bn + wn + l16;
        if (n < Cout) {
            float x = acc[v] + bias[n];
            out[(long long)m * Cout + n] = (_Float16)(x > 0.f ? x : 0.f);
        }
    }
}

// ------------------------------- pointwise ---------------------------------

__global__ void nchw_to_nhwc_k(const float* __restrict__ in, _Float16* __restrict__ out,
                               int B, int C, int H, int W) {
    long long i = (long long)blockIdx.x * blockDim.x + threadIdx.x;
    const long long total = (long long)B * C * H * W;
    if (i >= total) return;
    int c = (int)(i % C); long long r = i / C;
    int w = (int)(r % W); r /= W;
    int h = (int)(r % H); r /= H;
    int b = (int)r;
    out[i] = (_Float16)in[(((long long)b * C + c) * H + h) * W + w];
}

__global__ void nhwc_to_flat_chw_k(const _Float16* __restrict__ in, _Float16* __restrict__ out,
                                   int B, int H, int W, int C) {
    long long i = (long long)blockIdx.x * blockDim.x + threadIdx.x;
    const long long total = (long long)B * C * H * W;
    if (i >= total) return;
    int w = (int)(i % W); long long r = i / W;
    int h = (int)(r % H); r /= H;
    int c = (int)(r % C); r /= C;
    int b = (int)r;
    out[i] = in[(((long long)b * H + h) * W + w) * C + c];
}

__global__ void rsample_k(const float* __restrict__ mean, long long sm,
                          const float* __restrict__ stdv, long long ss,
                          const float* __restrict__ eps,  long long se,
                          _Float16* __restrict__ o1, long long so1,
                          float* __restrict__ o2, long long so2,
                          int rows, int cols) {
    int i = blockIdx.x * blockDim.x + threadIdx.x;
    if (i >= rows * cols) return;
    int r = i / cols, c = i % cols;
    float v = mean[r * sm + c] + stdv[r * ss + c] * eps[r * se + c];
    o1[r * so1 + c] = (_Float16)v;
    if (o2) o2[r * so2 + c] = v;
}

template <typename TA, typename TB, typename TO>
__global__ void concat2_k(const TA* __restrict__ a, long long sa, int ca,
                          const TB* __restrict__ b, long long sb, int cb,
                          TO* __restrict__ out, long long so, int rows) {
    const int cols = ca + cb;
    int i = blockIdx.x * blockDim.x + threadIdx.x;
    if (i >= rows * cols) return;
    int r = i / cols, c = i % cols;
    out[r * so + c] = (c < ca) ? (TO)a[r * sa + c] : (TO)b[r * sb + (c - ca)];
}

// torch.nn.GRUCell gates. gi = x@Wih+bih, gh = h@Whh+bhh, both [rows,3S] f32.
__global__ void gru_pw_k(const float* __restrict__ gi, const float* __restrict__ gh,
                         const _Float16* __restrict__ h, _Float16* __restrict__ hnew,
                         float* __restrict__ out2, long long so2, int rows, int S) {
    int i = blockIdx.x * blockDim.x + threadIdx.x;
    if (i >= rows * S) return;
    int r = i / S, c = i % S;
    const long long b3 = (long long)r * 3 * S;
    float ir = gi[b3 + c], iz = gi[b3 + S + c], in_ = gi[b3 + 2 * S + c];
    float hr = gh[b3 + c], hz = gh[b3 + S + c], hn  = gh[b3 + 2 * S + c];
    float rr = 1.f / (1.f + __expf(-(ir + hr)));
    float z  = 1.f / (1.f + __expf(-(iz + hz)));
    float n  = tanhf(in_ + rr * hn);
    float hv = (1.f - z) * n + z * (float)h[(long long)r * S + c];
    hnew[(long long)r * S + c] = (_Float16)hv;
    out2[r * so2 + c] = hv;
}

// ------------------------------- host side ---------------------------------

static inline size_t pack_elems(int K, int N) {
    return (size_t)((K + 31) / 32) * (size_t)(((N + 31) / 32) * 2) * 512;
}
static void launch_pack(const float* src, _Float16* dst, int K, int N, int transposed,
                        int convCin, hipStream_t st) {
    long long total = (long long)pack_elems(K, N);
    pack_wsw_k<<<dim3((unsigned)((total + 255) / 256)), 256, 0, st>>>(
        src, dst, K, N, transposed, convCin);
}

template <typename TC>
static void launch_gemm(int act, const _Float16* A, long long sA, const _Float16* Bsw,
                        const float* bias, TC* C, long long sC,
                        int M, int N, int K, hipStream_t st) {
    dim3 g((N + 31) / 32, (M + 31) / 32), blk(128);
    switch (act) {
        case ACT_RELU: wmma_gemm<ACT_RELU, TC><<<g, blk, 0, st>>>(A, sA, Bsw, bias, C, sC, N, K); break;
        case ACT_SP:   wmma_gemm<ACT_SP,   TC><<<g, blk, 0, st>>>(A, sA, Bsw, bias, C, sC, N, K); break;
        default:       wmma_gemm<ACT_NONE, TC><<<g, blk, 0, st>>>(A, sA, Bsw, bias, C, sC, N, K); break;
    }
}

static void launch_conv(const _Float16* in, const _Float16* Wsw, const float* b,
                        _Float16* out, int Bn, int Cin, int H, int W,
                        int Cout, int OH, int OW, hipStream_t st) {
    const int M = Bn * OH * OW;
    dim3 g((Cout + 31) / 32, M / 32), blk(128);
    wmma_conv<<<g, blk, 0, st>>>(in, Wsw, b, out, Cin, H, W, Cout, OH, OW);
}

struct StochP {   // packed trunk/mean/std weights + f32 biases
    const _Float16 *tw1, *tw2, *mw, *sw;
    const float *tb1, *tb2, *mb, *sb;
    int ip, op;
};

static void run_stochastic(const StochP& p, const _Float16* x, long long sx,
                           _Float16* t1, _Float16* t2,
                           float* mOut, long long smo, float* sOut, long long sso,
                           hipStream_t st) {
    launch_gemm<_Float16>(ACT_RELU, x,  sx,   p.tw1, p.tb1, t1, 1024, 32, 1024, p.ip, st);
    launch_gemm<_Float16>(ACT_RELU, t1, 1024, p.tw2, p.tb2, t2, 1024, 32, 1024, 1024, st);
    launch_gemm<float>(ACT_NONE, t2, 1024, p.mw, p.mb, mOut, smo, 32, p.op, 1024, st);
    launch_gemm<float>(ACT_SP,   t2, 1024, p.sw, p.sb, sOut, sso, 32, p.op, 1024, st);
}

extern "C" void kernel_launch(void* const* d_in, const int* in_sizes, int n_in,
                              void* d_out, int out_size, void* d_ws, size_t ws_size,
                              hipStream_t stream) {
    // ---- inputs (setup_inputs order; params pytree flattened, keys sorted) ----
    const float* obs     = (const float*)d_in[0];   // [32,32,3,64,64]
    const float* actions = (const float*)d_in[1];   // [32,31,6]
    const float* cat_w1  = (const float*)d_in[2];   const float* cat_b1 = (const float*)d_in[3];
    const float* cat_w2  = (const float*)d_in[4];   const float* cat_b2 = (const float*)d_in[5];
    const float* c1w = (const float*)d_in[6];  const float* c1b = (const float*)d_in[7];
    const float* c2w = (const float*)d_in[8];  const float* c2b = (const float*)d_in[9];
    const float* c3w = (const float*)d_in[10]; const float* c3b = (const float*)d_in[11];
    const float* c4w = (const float*)d_in[12]; const float* c4b = (const float*)d_in[13];
    const float* fcw = (const float*)d_in[14]; const float* fcb = (const float*)d_in[15];
    const float* gru_bhh = (const float*)d_in[16];
    const float* gru_bih = (const float*)d_in[17];
    const float* gru_whh = (const float*)d_in[18];  // [200,600]
    const float* gru_wih = (const float*)d_in[19];  // [200,600]
    const float* eps_init  = (const float*)d_in[44];  // [32,200]
    const float* eps_prior = (const float*)d_in[45];  // [31,32,30]
    float* out = (float*)d_out;                       // [32,32,320]
    (void)in_sizes; (void)n_in; (void)out_size; (void)ws_size;

    // ---- workspace ----
    char* base = (char*)d_ws;
    size_t off = 0;
    auto alloc = [&](size_t bytes) -> void* {
        void* p = base + off; off += (bytes + 255) & ~(size_t)255; return p;
    };
    auto ah = [&](size_t elems) { return (_Float16*)alloc(elems * sizeof(_Float16)); };
    auto af = [&](size_t elems) { return (float*)alloc(elems * sizeof(float)); };

    _Float16* obs_h = ah(1024ull * 64 * 64 * 3);
    _Float16* a1    = ah(1024ull * 31 * 31 * 32);
    _Float16* a2    = ah(1024ull * 14 * 14 * 64);
    _Float16* a3    = ah(1024ull * 6 * 6 * 128);
    _Float16* a4    = ah(1024ull * 2 * 2 * 256);
    _Float16* a4c   = ah(1024ull * 1024);
    _Float16* en    = ah(1024ull * 1024);   // [b = n*32+t][1024]
    _Float16* t1    = ah(32 * 1024);
    _Float16* t2    = ah(32 * 1024);
    _Float16* state = ah(32 * 200);
    _Float16* cc    = ah(32 * 1224);
    _Float16* xcat  = ah(32 * 40);          // stride 40: rows 16B-aligned
    _Float16* xb    = ah(32 * 200);
    _Float16* samp  = ah(32 * 30);
    float* m0 = af(32 * 200);
    float* s0 = af(32 * 200);
    float* gi = af(32 * 600);
    float* gh = af(32 * 600);

    // packed (fragment-ordered) f16 weights
    _Float16* p_c1  = ah(pack_elems(48, 32));
    _Float16* p_c2  = ah(pack_elems(512, 64));
    _Float16* p_c3  = ah(pack_elems(1024, 128));
    _Float16* p_c4  = ah(pack_elems(2048, 256));
    _Float16* p_fc  = ah(pack_elems(1024, 1024));
    _Float16* p_ct1 = ah(pack_elems(36, 1024));
    _Float16* p_ct2 = ah(pack_elems(1024, 200));
    _Float16* p_wih = ah(pack_elems(200, 600));
    _Float16* p_whh = ah(pack_elems(200, 600));
    _Float16* p_it1 = ah(pack_elems(1024, 1024));
    _Float16* p_it2 = ah(pack_elems(1024, 1024));
    _Float16* p_im  = ah(pack_elems(1024, 200));
    _Float16* p_is  = ah(pack_elems(1024, 200));
    _Float16* p_pt1 = ah(pack_elems(1224, 1024));
    _Float16* p_pt2 = ah(pack_elems(1024, 1024));
    _Float16* p_pm  = ah(pack_elems(1024, 30));
    _Float16* p_ps  = ah(pack_elems(1024, 30));
    _Float16* p_rt1 = ah(pack_elems(200, 1024));
    _Float16* p_rt2 = ah(pack_elems(1024, 1024));
    _Float16* p_rm  = ah(pack_elems(1024, 30));
    _Float16* p_rs  = ah(pack_elems(1024, 30));

    // ---- weight packing (conv weights [N][K]: transposed=1, channel-minor K) ----
    launch_pack(c1w, p_c1, 48, 32, 1, 3, stream);
    launch_pack(c2w, p_c2, 512, 64, 1, 32, stream);
    launch_pack(c3w, p_c3, 1024, 128, 1, 64, stream);
    launch_pack(c4w, p_c4, 2048, 256, 1, 128, stream);
    launch_pack(fcw, p_fc, 1024, 1024, 0, 0, stream);
    launch_pack(cat_w1, p_ct1, 36, 1024, 0, 0, stream);
    launch_pack(cat_w2, p_ct2, 1024, 200, 0, 0, stream);
    launch_pack(gru_wih, p_wih, 200, 600, 0, 0, stream);
    launch_pack(gru_whh, p_whh, 200, 600, 0, 0, stream);
    launch_pack((const float*)d_in[24], p_it1, 1024, 1024, 0, 0, stream);
    launch_pack((const float*)d_in[26], p_it2, 1024, 1024, 0, 0, stream);
    launch_pack((const float*)d_in[20], p_im, 1024, 200, 0, 0, stream);
    launch_pack((const float*)d_in[22], p_is, 1024, 200, 0, 0, stream);
    launch_pack((const float*)d_in[32], p_pt1, 1224, 1024, 0, 0, stream);
    launch_pack((const float*)d_in[34], p_pt2, 1024, 1024, 0, 0, stream);
    launch_pack((const float*)d_in[28], p_pm, 1024, 30, 0, 0, stream);
    launch_pack((const float*)d_in[30], p_ps, 1024, 30, 0, 0, stream);
    launch_pack((const float*)d_in[40], p_rt1, 200, 1024, 0, 0, stream);
    launch_pack((const float*)d_in[42], p_rt2, 1024, 1024, 0, 0, stream);
    launch_pack((const float*)d_in[36], p_rm, 1024, 30, 0, 0, stream);
    launch_pack((const float*)d_in[38], p_rs, 1024, 30, 0, 0, stream);

    StochP initS  = {p_it1, p_it2, p_im, p_is,
                     (const float*)d_in[25], (const float*)d_in[27],
                     (const float*)d_in[21], (const float*)d_in[23], 1024, 200};
    StochP postS  = {p_pt1, p_pt2, p_pm, p_ps,
                     (const float*)d_in[33], (const float*)d_in[35],
                     (const float*)d_in[29], (const float*)d_in[31], 1224, 30};
    StochP priorS = {p_rt1, p_rt2, p_rm, p_rs,
                     (const float*)d_in[41], (const float*)d_in[43],
                     (const float*)d_in[37], (const float*)d_in[39], 200, 30};

    // ---- encoder ----
    {
        long long tot = 1024ll * 3 * 64 * 64;
        nchw_to_nhwc_k<<<dim3((unsigned)((tot + 255) / 256)), 256, 0, stream>>>(
            obs, obs_h, 1024, 3, 64, 64);
    }
    launch_conv(obs_h, p_c1, c1b, a1, 1024,   3, 64, 64,  32, 31, 31, stream);
    launch_conv(a1,    p_c2, c2b, a2, 1024,  32, 31, 31,  64, 14, 14, stream);
    launch_conv(a2,    p_c3, c3b, a3, 1024,  64, 14, 14, 128,  6,  6, stream);
    launch_conv(a3,    p_c4, c4b, a4, 1024, 128,  6,  6, 256,  2,  2, stream);
    {
        long long tot = 1024ll * 256 * 2 * 2;
        nhwc_to_flat_chw_k<<<dim3((unsigned)((tot + 255) / 256)), 256, 0, stream>>>(
            a4, a4c, 1024, 2, 2, 256);
    }
    launch_gemm<_Float16>(ACT_NONE, a4c, 1024, p_fc, fcb, en, 1024, 1024, 1024, 1024, stream);

    const long long EN_ROW = 32ll * 1024;  // en[:, t] row n at (n*32+t)*1024

    // ---- t = 0 ----
    run_stochastic(initS, en, EN_ROW, t1, t2, m0, 200, s0, 200, stream);
    rsample_k<<<(32 * 200 + 255) / 256, 256, 0, stream>>>(
        m0, 200, s0, 200, eps_init, 200, state, 200, out, 320, 32, 200);
    run_stochastic(priorS, state, 200, t1, t2, out + 200, 320, out + 230, 320, stream);
    concat2_k<_Float16, _Float16, _Float16><<<(32 * 1224 + 255) / 256, 256, 0, stream>>>(
        state, 200, 200, en, EN_ROW, 1024, cc, 1224, 32);
    run_stochastic(postS, cc, 1224, t1, t2, out + 260, 320, out + 290, 320, stream);

    // ---- sequential scan t = 1..31 ----
    for (int t = 1; t < 32; ++t) {
        const int i = t - 1;
        float* outp = out + (size_t)t * 32 * 320;
        float* outprev = out + (size_t)(t - 1) * 32 * 320;

        rsample_k<<<(32 * 30 + 255) / 256, 256, 0, stream>>>(
            outprev + 200, 320, outprev + 230, 320,
            eps_prior + (size_t)i * 32 * 30, 30, samp, 30, nullptr, 0, 32, 30);

        concat2_k<_Float16, float, _Float16><<<(32 * 36 + 255) / 256, 256, 0, stream>>>(
            samp, 30, 30, actions + (size_t)i * 6, 31 * 6, 6, xcat, 40, 32);

        launch_gemm<_Float16>(ACT_RELU, xcat, 40,  p_ct1, cat_b1, t1, 1024, 32, 1024, 36, stream);
        launch_gemm<_Float16>(ACT_NONE, t1, 1024,  p_ct2, cat_b2, xb, 200, 32, 200, 1024, stream);

        launch_gemm<float>(ACT_NONE, xb,    200, p_wih, gru_bih, gi, 600, 32, 600, 200, stream);
        launch_gemm<float>(ACT_NONE, state, 200, p_whh, gru_bhh, gh, 600, 32, 600, 200, stream);
        gru_pw_k<<<(32 * 200 + 255) / 256, 256, 0, stream>>>(
            gi, gh, state, state, outp, 320, 32, 200);

        run_stochastic(priorS, state, 200, t1, t2, outp + 200, 320, outp + 230, 320, stream);
        concat2_k<_Float16, _Float16, _Float16><<<(32 * 1224 + 255) / 256, 256, 0, stream>>>(
            state, 200, 200, en + (size_t)t * 1024, EN_ROW, 1024, cc, 1224, 32);
        run_stochastic(postS, cc, 1224, t1, t2, outp + 260, 320, outp + 290, 320, stream);
    }
}
